// BasedLinearAttention_65601330479114
// MI455X (gfx1250) — compile-verified, matrix operand order checked by reading, not run on hardware
//
#include <hip/hip_runtime.h>
#include <hip/hip_bf16.h>

typedef __attribute__((ext_vector_type(16))) __bf16 v16bf;
typedef __attribute__((ext_vector_type(8)))  __bf16 v8bf;
typedef __attribute__((ext_vector_type(8)))  float  v8f;

#define WMMA_BF16(a, b, c) \
  __builtin_amdgcn_wmma_f32_16x16x32_bf16(false, (a), false, (b), (short)0, (c), false, false)

// Load a 16x32 bf16 A-fragment (or B^T fragment, identical layout) for wave32.
// Per CDNA5 ISA 7.12.2: lane L (L&15 selects row), klo = (L&16)?8:0 selects the
// K-half; each lane holds K = [klo..klo+7] and [16+klo..23+klo] of its row.
// p must point at &Mrow_base[row][k + klo] of a K-major matrix, 16B aligned.
__device__ __forceinline__ v16bf frag_ld(const __bf16* p) {
  v8bf lo = *(const v8bf*)(p);
  v8bf hi = *(const v8bf*)(p + 16);
  return __builtin_shufflevector(lo, hi, 0, 1, 2, 3, 4, 5, 6, 7,
                                 8, 9, 10, 11, 12, 13, 14, 15);
}

__global__ void cvt_f32_bf16(const float* __restrict__ in, __bf16* __restrict__ out, int n) {
  int i = blockIdx.x * blockDim.x + threadIdx.x;
  int stride = gridDim.x * blockDim.x;
  for (; i < n; i += stride) out[i] = (__bf16)in[i];
}

// C[M,N] = A[M,K] * B[N,K]^T.  A,B bf16 K-major.  128x128 block, 8 waves,
// each wave 32(M) x 64(N) = 8 accum tiles.  Operands stream from L2 (192MB).
template <bool STORE_F32>
__global__ __launch_bounds__(256) void gemm_wmma(const __bf16* __restrict__ A,
                                                 const __bf16* __restrict__ B,
                                                 void* __restrict__ Cout,
                                                 int M, int N, int K) {
  const int lane = threadIdx.x & 31;
  const int wave = threadIdx.x >> 5;
  const int l15  = lane & 15;
  const int klo  = (lane & 16) ? 8 : 0;
  const int hi8  = (lane & 16) ? 8 : 0;
  const int mbase = blockIdx.y * 128 + (wave & 3) * 32;
  const int nbase = blockIdx.x * 128 + (wave >> 2) * 64;

  const v8f zf = {0.f, 0.f, 0.f, 0.f, 0.f, 0.f, 0.f, 0.f};
  v8f acc[2][4];
#pragma unroll
  for (int i = 0; i < 2; ++i)
#pragma unroll
    for (int j = 0; j < 4; ++j) acc[i][j] = zf;

  for (int k = 0; k < K; k += 32) {
    v16bf a0 = frag_ld(A + (size_t)(mbase + l15) * K + k + klo);
    v16bf a1 = frag_ld(A + (size_t)(mbase + 16 + l15) * K + k + klo);
#pragma unroll
    for (int j = 0; j < 4; ++j) {
      v16bf bf = frag_ld(B + (size_t)(nbase + 16 * j + l15) * K + k + klo);
      acc[0][j] = WMMA_BF16(a0, bf, acc[0][j]);
      acc[1][j] = WMMA_BF16(a1, bf, acc[1][j]);
    }
  }
  // C layout: lane holds col N = l15, rows M = r + hi8 (r = reg 0..7)
#pragma unroll
  for (int i = 0; i < 2; ++i)
#pragma unroll
    for (int j = 0; j < 4; ++j)
#pragma unroll
      for (int r = 0; r < 8; ++r) {
        const size_t row = mbase + 16 * i + r + hi8;
        const size_t col = nbase + 16 * j + l15;
        float v = acc[i][j][r];
        if constexpr (STORE_F32)
          ((float*)Cout)[row * N + col] = v;
        else
          ((__bf16*)Cout)[row * N + col] = (__bf16)v;
      }
}

// qkv[4096,1920] -> qpad/kpad [b,h,t,32] (upper 16 K zero-padded), vT [b,h,128,t]
__global__ void repack_qkv(const __bf16* __restrict__ qkv,
                           __bf16* __restrict__ qpad, __bf16* __restrict__ kpad,
                           __bf16* __restrict__ vT) {
  const int row = blockIdx.x;      // 0..4095
  const int b = row >> 11;
  const int t = row & 2047;
  for (int c = threadIdx.x; c < 1920 + 384; c += blockDim.x) {
    if (c < 192) {
      int h = c >> 4, i = c & 15;
      qpad[(((size_t)(b * 12 + h)) * 2048 + t) * 32 + i] = qkv[(size_t)row * 1920 + c];
    } else if (c < 384) {
      int cc = c - 192;
      int h = cc >> 4, i = cc & 15;
      kpad[(((size_t)(b * 12 + h)) * 2048 + t) * 32 + i] = qkv[(size_t)row * 1920 + c];
    } else if (c < 1920) {
      int cc = c - 384;
      int h = cc >> 7, dvi = cc & 127;
      vT[(((size_t)(b * 12 + h)) * 128 + dvi) * 2048 + t] = qkv[(size_t)row * 1920 + c];
    } else {  // zero the K-pad region
      int cc = c - 1920;
      int sel = cc / 192, rem = cc % 192;
      int h = rem >> 4, i = 16 + (rem & 15);
      __bf16* dst = sel ? kpad : qpad;
      dst[(((size_t)(b * 12 + h)) * 2048 + t) * 32 + i] = (__bf16)0.f;
    }
  }
}

// Causal "based" linear attention via quadratic form + scalar kernel:
// S = 1 + a/4 + a^2/32, a = q.k (d=16).  One block per (b,h,64-row q tile).
// 4 waves, each owns 16 query rows and the full dv=128 accumulators.
// Causal mask is hoisted: only the final (diagonal) key block masks.
__global__ __launch_bounds__(128) void based_attn(const __bf16* __restrict__ qpad,
                                                  const __bf16* __restrict__ kpad,
                                                  const __bf16* __restrict__ vT,
                                                  __bf16* __restrict__ y) {
  const int qtile = blockIdx.x;  // 0..31
  const int h = blockIdx.y, b = blockIdx.z;
  const int lane = threadIdx.x & 31;
  const int w = threadIdx.x >> 5;
  const int l15 = lane & 15;
  const int klo = (lane & 16) ? 8 : 0;
  const int hi8 = (lane & 16) ? 8 : 0;

  __shared__ __bf16 sS[4][16 * 64];  // per-wave 16x64 S staging (C-layout -> A-frag transpose)

  const size_t bh = (size_t)b * 12 + h;
  const __bf16* qb = qpad + bh * 2048 * 32;
  const __bf16* kb = kpad + bh * 2048 * 32;
  const __bf16* vb = vT + bh * 128 * 2048;

  const int q0 = qtile * 64 + w * 16;  // t-index of this wave's first query row
  const v16bf qa = frag_ld(qb + (size_t)(q0 + l15) * 32 + klo);  // loaded once

  const v8f zf = {0.f, 0.f, 0.f, 0.f, 0.f, 0.f, 0.f, 0.f};
  v8f num[8];
#pragma unroll
  for (int n = 0; n < 8; ++n) num[n] = zf;
  float den[8] = {0.f, 0.f, 0.f, 0.f, 0.f, 0.f, 0.f, 0.f};

  // ---------- steady state: strictly-below-diagonal key blocks, no masking ----------
  for (int kblk = 0; kblk < qtile; ++kblk) {
    const int s0 = kblk * 64;
#pragma unroll
    for (int j = 0; j < 4; ++j) {
      v16bf kf = frag_ld(kb + (size_t)(s0 + 16 * j + l15) * 32 + klo);
      v8f s = WMMA_BF16(qa, kf, zf);
#pragma unroll
      for (int r = 0; r < 8; ++r) {
        float a = s[r];
        float val = 1.0f + 0.25f * a + a * a * (1.0f / 32.0f);
        den[r] += val;
        sS[w][(r + hi8) * 64 + 16 * j + l15] = (__bf16)val;
      }
    }
#pragma unroll
    for (int kk = 0; kk < 2; ++kk) {
      const __bf16* sp = &sS[w][l15 * 64 + kk * 32 + klo];
      v8bf lo = *(const v8bf*)sp;
      v8bf hi = *(const v8bf*)(sp + 16);
      v16bf sa = __builtin_shufflevector(lo, hi, 0, 1, 2, 3, 4, 5, 6, 7,
                                         8, 9, 10, 11, 12, 13, 14, 15);
#pragma unroll
      for (int n = 0; n < 8; ++n) {
        v16bf vf = frag_ld(vb + (size_t)(16 * n + l15) * 2048 + s0 + kk * 32 + klo);
        num[n] = WMMA_BF16(sa, vf, num[n]);
      }
    }
  }

  // ---------- diagonal key block: j<w full, j==w triangular, j>w all-zero ----------
  {
    const int s0 = qtile * 64;
#pragma unroll
    for (int j = 0; j < 4; ++j) {
      if (j > w) {  // entirely above diagonal for this wave: zero-fill, skip WMMA
#pragma unroll
        for (int r = 0; r < 8; ++r)
          sS[w][(r + hi8) * 64 + 16 * j + l15] = (__bf16)0.f;
        continue;
      }
      v16bf kf = frag_ld(kb + (size_t)(s0 + 16 * j + l15) * 32 + klo);
      v8f s = WMMA_BF16(qa, kf, zf);
#pragma unroll
      for (int r = 0; r < 8; ++r) {
        float a = s[r];
        float val = 1.0f + 0.25f * a + a * a * (1.0f / 32.0f);
        // triangular tile: key col l15 vs query row r+hi8 (lane-constant compare)
        if (j == w && l15 > r + hi8) val = 0.0f;
        den[r] += val;
        sS[w][(r + hi8) * 64 + 16 * j + l15] = (__bf16)val;
      }
    }
    // S columns beyond 16*(w+1) are zero -> clip the K-range of S @ V
    const int kkmax = w >> 1;  // kk=0 always; kk=1 only for waves 2,3
#pragma unroll
    for (int kk = 0; kk < 2; ++kk) {
      if (kk > kkmax) break;
      const __bf16* sp = &sS[w][l15 * 64 + kk * 32 + klo];
      v8bf lo = *(const v8bf*)sp;
      v8bf hi = *(const v8bf*)(sp + 16);
      v16bf sa = __builtin_shufflevector(lo, hi, 0, 1, 2, 3, 4, 5, 6, 7,
                                         8, 9, 10, 11, 12, 13, 14, 15);
#pragma unroll
      for (int n = 0; n < 8; ++n) {
        v16bf vf = frag_ld(vb + (size_t)(16 * n + l15) * 2048 + s0 + kk * 32 + klo);
        num[n] = WMMA_BF16(sa, vf, num[n]);
      }
    }
  }

  // den: lane holds partial sum over its 16 key columns; reduce across the
  // 16-lane half (xor masks 1,2,4,8 stay within each half of the wave).
  float dfin[8];
#pragma unroll
  for (int r = 0; r < 8; ++r) {
    float d = den[r];
    d += __shfl_xor(d, 1, 32);
    d += __shfl_xor(d, 2, 32);
    d += __shfl_xor(d, 4, 32);
    d += __shfl_xor(d, 8, 32);
    dfin[r] = d + 1e-12f;
  }

  // y[b*2048 + t][h*128 + dv]  (bf16, feeds the Wo GEMM)
#pragma unroll
  for (int n = 0; n < 8; ++n)
#pragma unroll
    for (int r = 0; r < 8; ++r) {
      float v = num[n][r] / dfin[r];
      const size_t row = (size_t)b * 2048 + q0 + r + hi8;
      const size_t col = (size_t)h * 128 + 16 * n + l15;
      y[row * 1536 + col] = (__bf16)v;
    }
}

extern "C" void kernel_launch(void* const* d_in, const int* in_sizes, int n_in,
                              void* d_out, int out_size, void* d_ws, size_t ws_size,
                              hipStream_t stream) {
  (void)in_sizes; (void)n_in; (void)out_size; (void)ws_size;
  const float* x  = (const float*)d_in[0];
  const float* Wq = (const float*)d_in[1];
  const float* Wk = (const float*)d_in[2];
  const float* Wv = (const float*)d_in[3];
  const float* Wo = (const float*)d_in[4];

  const int B = 2, T = 2048, HID = 1536, H = 12;
  const int M = B * T;              // 4096
  const int NQKV = 1920;            // 192 q + 192 k + 1536 v

  char* ws = (char*)d_ws;
  size_t off = 0;
  auto alloc = [&](size_t bytes) {
    void* p = ws + off;
    off += (bytes + 255) & ~(size_t)255;
    return p;
  };
  __bf16* xb   = (__bf16*)alloc((size_t)M * HID * 2);
  __bf16* Wqkv = (__bf16*)alloc((size_t)NQKV * HID * 2);
  __bf16* Wob  = (__bf16*)alloc((size_t)HID * HID * 2);
  __bf16* qkv  = (__bf16*)alloc((size_t)M * NQKV * 2);
  __bf16* qpad = (__bf16*)alloc((size_t)B * H * T * 32 * 2);
  __bf16* kpad = (__bf16*)alloc((size_t)B * H * T * 32 * 2);
  __bf16* vT   = (__bf16*)alloc((size_t)B * H * 128 * T * 2);
  __bf16* yb   = (__bf16*)alloc((size_t)M * HID * 2);

  cvt_f32_bf16<<<512, 256, 0, stream>>>(x, xb, M * HID);
  cvt_f32_bf16<<<128, 256, 0, stream>>>(Wq, Wqkv, 192 * HID);
  cvt_f32_bf16<<<128, 256, 0, stream>>>(Wk, Wqkv + (size_t)192 * HID, 192 * HID);
  cvt_f32_bf16<<<512, 256, 0, stream>>>(Wv, Wqkv + (size_t)384 * HID, 1536 * HID);
  cvt_f32_bf16<<<512, 256, 0, stream>>>(Wo, Wob, HID * HID);

  gemm_wmma<false><<<dim3(NQKV / 128, M / 128), 256, 0, stream>>>(xb, Wqkv, qkv, M, NQKV, HID);
  repack_qkv<<<M, 256, 0, stream>>>(qkv, qpad, kpad, vT);
  based_attn<<<dim3(T / 64, H, B), 128, 0, stream>>>(qpad, kpad, vT, yb);
  gemm_wmma<true><<<dim3(HID / 128, M / 128), 256, 0, stream>>>(yb, Wob, d_out, M, HID, HID);
}
